// GPTBase_64536178590124
// MI455X (gfx1250) — matrix operand order repaired and checked by reading, compiled, not run on hardware
//
#include <hip/hip_runtime.h>

// ---------------------------------------------------------------------------
// MoE GPT forward (B=4,T=1024,D=1024,H=16,L=2,E=8,DFF=4096,V=50257) for
// MI455X / gfx1250.  All GEMMs run on the matrix pipe via
// v_wmma_f32_16x16x32_bf16 (f32 data converted to bf16 while staging to LDS,
// f32 accumulation).  Attention is a flash-style streaming-softmax kernel
// (also WMMA).  RMSNorm uses the gfx1250 async global->LDS copy path when the
// toolchain exposes it; GEMM/flash issue global_prefetch for next tiles.
// ---------------------------------------------------------------------------

#define Bsz 4
#define Tn  1024
#define Dm  1024
#define Hn  16
#define Ln  2
#define En  8
#define DFFn 4096
#define Vn  50257
#define HDm 64
#define NTOK (Bsz*Tn)   // 4096
#define TOPKn 1024

typedef __attribute__((ext_vector_type(16))) __bf16 v16bf;
typedef __attribute__((ext_vector_type(8)))  float  v8f;
typedef int v4i __attribute__((vector_size(16)));   // matches async-LDS builtin param

#define AS1 __attribute__((address_space(1)))
#define AS3 __attribute__((address_space(3)))

#if __has_builtin(__builtin_amdgcn_global_load_async_to_lds_b128)
#define HAVE_ASYNC_LDS 1
#else
#define HAVE_ASYNC_LDS 0
#endif

static __device__ __forceinline__ void wait_asynccnt0() {
#if __has_builtin(__builtin_amdgcn_s_wait_asynccnt)
    __builtin_amdgcn_s_wait_asynccnt(0);
#else
    asm volatile("s_wait_asynccnt 0" ::: "memory");
#endif
}

union Frag {
    v16bf v;
    uint4 q[2];
};

static __device__ __forceinline__ unsigned short f2bfu(float a) {
    unsigned int u = __float_as_uint(a);
    return (unsigned short)((u + 0x7FFFu + ((u >> 16) & 1u)) >> 16);
}
static __device__ __forceinline__ unsigned int bf2(float a, float b) {
    return (unsigned int)f2bfu(a) | ((unsigned int)f2bfu(b) << 16);
}
static __device__ __forceinline__ uint4 pack8(float4 a, float4 b) {
    uint4 r;
    r.x = bf2(a.x, a.y); r.y = bf2(a.z, a.w);
    r.z = bf2(b.x, b.y); r.w = bf2(b.z, b.w);
    return r;
}
static __device__ __forceinline__ v8f zero8() {
    v8f z;
#pragma unroll
    for (int i = 0; i < 8; ++i) z[i] = 0.0f;
    return z;
}

// ---------------------------------------------------------------------------
// Generic GEMM: C[M,N] = act( A[M,K] * W[N,K]^T  (+ C if ACC_) )
// A,W,C are f32 row-major; bf16 WMMA compute, f32 accumulate.
// Tile 128x128x64, 256 threads = 8 wave32 waves laid out 4(M) x 2(N).
// Batched over blockIdx.z with element strides sA/sW/sC.
// ---------------------------------------------------------------------------
template<int GELU_, int ACC_>
__global__ __launch_bounds__(256)
void gemm_xwt(const float* __restrict__ A, long long sA,
              const float* __restrict__ W, long long sW,
              float* __restrict__ C, long long sC,
              int M, int N, int K, int lda, int ldb, int ldc)
{
    __shared__ unsigned short As[128 * 72];
    __shared__ unsigned short Bs[128 * 72];
    const int tid = threadIdx.x;
    A += (long long)blockIdx.z * sA;
    W += (long long)blockIdx.z * sW;
    C += (long long)blockIdx.z * sC;
    const int m0 = blockIdx.y * 128, n0 = blockIdx.x * 128;
    const int lane = tid & 31, wid = tid >> 5;
    const int l15 = lane & 15, khalf = lane >> 4;
    const int wm = (wid & 3) * 32, wn = (wid >> 2) * 64;

    v8f acc[2][4];
#pragma unroll
    for (int mi = 0; mi < 2; ++mi)
#pragma unroll
        for (int ni = 0; ni < 4; ++ni) acc[mi][ni] = zero8();

    for (int k0 = 0; k0 < K; k0 += 64) {
        // stage A and W tiles (f32 -> bf16) into LDS, row stride 72 (16B-aligned)
#pragma unroll
        for (int i = 0; i < 4; ++i) {
            int id8 = tid + i * 256;            // 1024 chunks of 8 floats
            int row = id8 >> 3, c8 = (id8 & 7) * 8;
            float4 f0 = {0, 0, 0, 0}, f1 = {0, 0, 0, 0};
            int gm = m0 + row;
            if (gm < M) {
                const float* p = A + (long long)gm * lda + k0 + c8;
                f0 = *(const float4*)p; f1 = *(const float4*)(p + 4);
            }
            *(uint4*)&As[row * 72 + c8] = pack8(f0, f1);
            float4 g0 = {0, 0, 0, 0}, g1 = {0, 0, 0, 0};
            int gn = n0 + row;
            if (gn < N) {
                const float* p = W + (long long)gn * ldb + k0 + c8;
                g0 = *(const float4*)p; g1 = *(const float4*)(p + 4);
            }
            *(uint4*)&Bs[row * 72 + c8] = pack8(g0, g1);
        }
        // prefetch next K-tile (hidden behind the WMMA loop below)
        if (k0 + 64 < K) {
            int r2 = tid >> 1, h2 = (tid & 1) * 32;
            int gm = m0 + r2, gn = n0 + r2;
            if (gm < M) __builtin_prefetch(A + (long long)gm * lda + k0 + 64 + h2, 0, 1);
            if (gn < N) __builtin_prefetch(W + (long long)gn * ldb + k0 + 64 + h2, 0, 1);
        }
        __syncthreads();
#pragma unroll
        for (int ks = 0; ks < 2; ++ks) {
            Frag a[2], bfrag[4];
#pragma unroll
            for (int mi = 0; mi < 2; ++mi) {
                const unsigned short* p = &As[(wm + mi * 16 + l15) * 72 + ks * 32 + khalf * 8];
                a[mi].q[0] = *(const uint4*)p;
                a[mi].q[1] = *(const uint4*)(p + 16);
            }
#pragma unroll
            for (int ni = 0; ni < 4; ++ni) {
                const unsigned short* p = &Bs[(wn + ni * 16 + l15) * 72 + ks * 32 + khalf * 8];
                bfrag[ni].q[0] = *(const uint4*)p;
                bfrag[ni].q[1] = *(const uint4*)(p + 16);
            }
#pragma unroll
            for (int mi = 0; mi < 2; ++mi)
#pragma unroll
                for (int ni = 0; ni < 4; ++ni)
                    acc[mi][ni] = __builtin_amdgcn_wmma_f32_16x16x32_bf16(
                        false, a[mi].v, false, bfrag[ni].v, (short)0, acc[mi][ni], false, false);
        }
        __syncthreads();
    }

#pragma unroll
    for (int mi = 0; mi < 2; ++mi)
#pragma unroll
        for (int ni = 0; ni < 4; ++ni)
#pragma unroll
            for (int r = 0; r < 8; ++r) {
                int m = m0 + wm + mi * 16 + khalf * 8 + r;
                int n = n0 + wn + ni * 16 + l15;
                if (m < M && n < N) {
                    float v = acc[mi][ni][r];
                    if (GELU_) v = 0.5f * v * (1.0f + erff(v * 0.70710678118f));
                    float* cp = C + (long long)m * ldc + n;
                    if (ACC_) *cp = *cp + v; else *cp = v;
                }
            }
}

// ---------------------------------------------------------------------------
// Flash attention: per (q-tile of 128 rows, b*H+h).  K/V streamed in 64-row
// tiles through LDS; S=Q*K^T and O+=P*V on the WMMA pipe; online softmax.
// Q is pre-scaled by 1/sqrt(HD)=0.125 at load.  V supplied transposed
// [B,H,HD,T] so it is directly the B=[N,K] GEMM operand.
// ---------------------------------------------------------------------------
__global__ __launch_bounds__(256)
void flash_attn(const float* __restrict__ Q, const float* __restrict__ Kg,
                const float* __restrict__ Vt, float* __restrict__ Y)
{
    __shared__ unsigned short Qs[128 * 72];
    __shared__ unsigned short Ks[64 * 72];
    __shared__ unsigned short Vs[64 * 72];
    __shared__ unsigned short Ps[128 * 72];
    __shared__ float red0[128], red1[128];
    __shared__ float mst[128], lst[128], msc[128];

    const int tid = threadIdx.x;
    const int lane = tid & 31, wid = tid >> 5;
    const int l15 = lane & 15, khalf = lane >> 4;
    const int wm = (wid & 3) * 32;        // q-row block of this wave
    const int wn = (wid >> 2) * 32;       // kv-col / hd-col block of this wave
    const int q0 = blockIdx.x * 128;
    const int bh = blockIdx.y;
    const int b = bh / Hn, h = bh % Hn;

    // load Q tile once, scaled
#pragma unroll
    for (int i = 0; i < 4; ++i) {
        int id8 = tid + i * 256;
        int row = id8 >> 3, c8 = (id8 & 7) * 8;
        const float* p = Q + ((long long)bh * Tn + q0 + row) * HDm + c8;
        float4 f0 = *(const float4*)p, f1 = *(const float4*)(p + 4);
        f0.x *= 0.125f; f0.y *= 0.125f; f0.z *= 0.125f; f0.w *= 0.125f;
        f1.x *= 0.125f; f1.y *= 0.125f; f1.z *= 0.125f; f1.w *= 0.125f;
        *(uint4*)&Qs[row * 72 + c8] = pack8(f0, f1);
    }
    if (tid < 128) { mst[tid] = -1e30f; lst[tid] = 0.0f; }

    v8f oacc[2][2];
#pragma unroll
    for (int mi = 0; mi < 2; ++mi)
#pragma unroll
        for (int ni = 0; ni < 2; ++ni) oacc[mi][ni] = zero8();

    const int jmax = (q0 + 127) >> 6;     // causal: only kv tiles <= diagonal
    for (int j = 0; j <= jmax; ++j) {
        const int kb = j * 64;
        // stage K tile [64 x HD] and Vt tile [HD x 64]
#pragma unroll
        for (int i = 0; i < 2; ++i) {
            int id8 = tid + i * 256;
            int row = id8 >> 3, c8 = (id8 & 7) * 8;
            const float* pk = Kg + ((long long)bh * Tn + kb + row) * HDm + c8;
            *(uint4*)&Ks[row * 72 + c8] = pack8(*(const float4*)pk, *(const float4*)(pk + 4));
            const float* pv = Vt + ((long long)bh * HDm + row) * Tn + kb + c8;
            *(uint4*)&Vs[row * 72 + c8] = pack8(*(const float4*)pv, *(const float4*)(pv + 4));
        }
        // prefetch next kv tile
        if (j < jmax) {
            int r2 = tid >> 1, h2 = (tid & 1) * 32;
            if (r2 < 64) {
                __builtin_prefetch(Kg + ((long long)bh * Tn + kb + 64 + r2) * HDm + h2, 0, 1);
            } else {
                __builtin_prefetch(Vt + ((long long)bh * HDm + (r2 - 64)) * Tn + kb + 64 + h2, 0, 1);
            }
        }
        __syncthreads();

        // S = Qs * Ks^T   (K-dim = HD = 64)
        v8f s[2][2];
#pragma unroll
        for (int mi = 0; mi < 2; ++mi)
#pragma unroll
            for (int ni = 0; ni < 2; ++ni) s[mi][ni] = zero8();
#pragma unroll
        for (int ks = 0; ks < 2; ++ks) {
            Frag a[2], bfrag[2];
#pragma unroll
            for (int mi = 0; mi < 2; ++mi) {
                const unsigned short* p = &Qs[(wm + mi * 16 + l15) * 72 + ks * 32 + khalf * 8];
                a[mi].q[0] = *(const uint4*)p;
                a[mi].q[1] = *(const uint4*)(p + 16);
            }
#pragma unroll
            for (int ni = 0; ni < 2; ++ni) {
                const unsigned short* p = &Ks[(wn + ni * 16 + l15) * 72 + ks * 32 + khalf * 8];
                bfrag[ni].q[0] = *(const uint4*)p;
                bfrag[ni].q[1] = *(const uint4*)(p + 16);
            }
#pragma unroll
            for (int mi = 0; mi < 2; ++mi)
#pragma unroll
                for (int ni = 0; ni < 2; ++ni)
                    s[mi][ni] = __builtin_amdgcn_wmma_f32_16x16x32_bf16(
                        false, a[mi].v, false, bfrag[ni].v, (short)0, s[mi][ni], false, false);
        }

        // causal mask on diagonal-overlapping tiles
        if (kb + 63 > q0) {
#pragma unroll
            for (int mi = 0; mi < 2; ++mi)
#pragma unroll
                for (int ni = 0; ni < 2; ++ni)
#pragma unroll
                    for (int r = 0; r < 8; ++r) {
                        int qg = q0 + wm + mi * 16 + khalf * 8 + r;
                        int kg = kb + wn + ni * 16 + l15;
                        if (kg > qg) s[mi][ni][r] = -1e30f;
                    }
        }

        // row max (cross-lane shuffle within 16 lanes, cross-wave via LDS)
#pragma unroll
        for (int mi = 0; mi < 2; ++mi)
#pragma unroll
            for (int r = 0; r < 8; ++r) {
                float v = fmaxf(s[mi][0][r], s[mi][1][r]);
                v = fmaxf(v, __shfl_xor(v, 1, 32));
                v = fmaxf(v, __shfl_xor(v, 2, 32));
                v = fmaxf(v, __shfl_xor(v, 4, 32));
                v = fmaxf(v, __shfl_xor(v, 8, 32));
                if (l15 == 0) {
                    int row = wm + mi * 16 + khalf * 8 + r;
                    if (wn) red1[row] = v; else red0[row] = v;
                }
            }
        __syncthreads();
        if (tid < 128) {
            float mo = mst[tid];
            float mn = fmaxf(mo, fmaxf(red0[tid], red1[tid]));
            float sc = __expf(mo - mn);
            msc[tid] = sc; lst[tid] *= sc; mst[tid] = mn;
        }
        __syncthreads();

        // P = exp(S-m) -> LDS bf16 A-tile; accumulate row sums
#pragma unroll
        for (int mi = 0; mi < 2; ++mi)
#pragma unroll
            for (int r = 0; r < 8; ++r) {
                int row = wm + mi * 16 + khalf * 8 + r;
                float mrow = mst[row];
                float p0 = __expf(s[mi][0][r] - mrow);
                float p1 = __expf(s[mi][1][r] - mrow);
                Ps[row * 72 + wn + l15]      = f2bfu(p0);
                Ps[row * 72 + wn + 16 + l15] = f2bfu(p1);
                float v = p0 + p1;
                v += __shfl_xor(v, 1, 32);
                v += __shfl_xor(v, 2, 32);
                v += __shfl_xor(v, 4, 32);
                v += __shfl_xor(v, 8, 32);
                if (l15 == 0) { if (wn) red1[row] = v; else red0[row] = v; }
            }
        __syncthreads();
        if (tid < 128) lst[tid] += red0[tid] + red1[tid];

        // rescale running O
#pragma unroll
        for (int mi = 0; mi < 2; ++mi)
#pragma unroll
            for (int r = 0; r < 8; ++r) {
                int row = wm + mi * 16 + khalf * 8 + r;
                float f = msc[row];
                oacc[mi][0][r] *= f;
                oacc[mi][1][r] *= f;
            }
        __syncthreads();

        // O += P * V   (K-dim = 64 kv)
#pragma unroll
        for (int ks = 0; ks < 2; ++ks) {
            Frag a[2], bfrag[2];
#pragma unroll
            for (int mi = 0; mi < 2; ++mi) {
                const unsigned short* p = &Ps[(wm + mi * 16 + l15) * 72 + ks * 32 + khalf * 8];
                a[mi].q[0] = *(const uint4*)p;
                a[mi].q[1] = *(const uint4*)(p + 16);
            }
#pragma unroll
            for (int ni = 0; ni < 2; ++ni) {
                const unsigned short* p = &Vs[(wn + ni * 16 + l15) * 72 + ks * 32 + khalf * 8];
                bfrag[ni].q[0] = *(const uint4*)p;
                bfrag[ni].q[1] = *(const uint4*)(p + 16);
            }
#pragma unroll
            for (int mi = 0; mi < 2; ++mi)
#pragma unroll
                for (int ni = 0; ni < 2; ++ni)
                    oacc[mi][ni] = __builtin_amdgcn_wmma_f32_16x16x32_bf16(
                        false, a[mi].v, false, bfrag[ni].v, (short)0, oacc[mi][ni], false, false);
        }
        __syncthreads();
    }

    // write O / l back merged as [B,T,D]
#pragma unroll
    for (int mi = 0; mi < 2; ++mi)
#pragma unroll
        for (int ni = 0; ni < 2; ++ni)
#pragma unroll
            for (int r = 0; r < 8; ++r) {
                int row = wm + mi * 16 + khalf * 8 + r;
                int col = wn + ni * 16 + l15;
                float l = fmaxf(lst[row], 1e-20f);
                Y[((long long)b * Tn + q0 + row) * Dm + h * HDm + col] =
                    oacc[mi][ni][r] / l;
            }
}

// ---------------------------------------------------------------------------
// Small support kernels
// ---------------------------------------------------------------------------
__global__ __launch_bounds__(256)
void embed_k(const int* __restrict__ idx, const float* __restrict__ wte,
             const float* __restrict__ wpe, float* __restrict__ x)
{
    int rowid = blockIdx.x;
    int t = rowid & (Tn - 1);
    long long id = idx[rowid];
    const float4* we = (const float4*)(wte + id * Dm);
    const float4* wp = (const float4*)(wpe + (long long)t * Dm);
    float4* o = (float4*)(x + (long long)rowid * Dm);
    for (int i = threadIdx.x; i < Dm / 4; i += 256) {
        float4 a = we[i], c = wp[i];
        float4 r; r.x = a.x + c.x; r.y = a.y + c.y; r.z = a.z + c.z; r.w = a.w + c.w;
        o[i] = r;
    }
}

// RMSNorm: stage the row into LDS (async global->LDS on gfx1250 when the
// builtin is available), then both passes (sumsq + normalize) read LDS.
__global__ __launch_bounds__(256)
void rmsnorm_k(const float* __restrict__ x, long long rowStride, long long rowOff,
               const float* __restrict__ w, float* __restrict__ o, int nrows)
{
    int row = blockIdx.x;
    if (row >= nrows) return;
    const float* xr = x + (long long)row * rowStride + rowOff;
    __shared__ float xs[Dm];
#if HAVE_ASYNC_LDS
    // 1024 floats = 4KB: each of the 256 threads async-copies 16B.
    __builtin_amdgcn_global_load_async_to_lds_b128(
        (AS1 v4i*)(xr + threadIdx.x * 4),
        (AS3 v4i*)&xs[threadIdx.x * 4], 0, 0);
    wait_asynccnt0();
    __syncthreads();
#else
    for (int i = threadIdx.x; i < Dm; i += 256) xs[i] = xr[i];
    __syncthreads();
#endif
    float ss = 0.0f;
    for (int i = threadIdx.x; i < Dm; i += 256) { float v = xs[i]; ss += v * v; }
    ss += __shfl_xor(ss, 1, 32);  ss += __shfl_xor(ss, 2, 32);
    ss += __shfl_xor(ss, 4, 32);  ss += __shfl_xor(ss, 8, 32);
    ss += __shfl_xor(ss, 16, 32);
    __shared__ float wsum[8];
    if ((threadIdx.x & 31) == 0) wsum[threadIdx.x >> 5] = ss;
    __syncthreads();
    float tot = 0.0f;
#pragma unroll
    for (int i = 0; i < 8; ++i) tot += wsum[i];
    float r = rsqrtf(tot * (1.0f / Dm) + 1e-6f);
    for (int i = threadIdx.x; i < Dm; i += 256)
        o[(long long)row * Dm + i] = xs[i] * r * w[i];
}

__global__ __launch_bounds__(256)
void split_qkv_k(const float* __restrict__ qkv, float* __restrict__ Q,
                 float* __restrict__ K, float* __restrict__ Vt)
{
    int rowid = blockIdx.x;
    int b = rowid / Tn, t = rowid & (Tn - 1);
    const float* src = qkv + (long long)rowid * 3 * Dm;
    for (int i = threadIdx.x; i < Dm; i += 256) {
        int h = i >> 6, hd = i & 63;
        long long bh = (long long)b * Hn + h;
        Q[(bh * Tn + t) * HDm + hd] = src[i];
        K[(bh * Tn + t) * HDm + hd] = src[Dm + i];
        Vt[(bh * HDm + hd) * Tn + t] = src[2 * Dm + i];
    }
}

__global__ __launch_bounds__(256)
void router_softmax_k(const float* __restrict__ rl, float* __restrict__ probsT)
{
    int n = blockIdx.x * 256 + threadIdx.x;
    if (n >= NTOK) return;
    float v[En];
    float mx = -1e30f;
#pragma unroll
    for (int e = 0; e < En; ++e) { v[e] = rl[(long long)n * En + e]; mx = fmaxf(mx, v[e]); }
    float s = 0.0f;
#pragma unroll
    for (int e = 0; e < En; ++e) { v[e] = __expf(v[e] - mx); s += v[e]; }
    float inv = 1.0f / s;
#pragma unroll
    for (int e = 0; e < En; ++e) probsT[(long long)e * NTOK + n] = v[e] * inv;
}

// expert-choice top-k: per expert, bitonic-sort all 4096 (prob, token) pairs
// in LDS, keep the 1024 largest.
__global__ __launch_bounds__(1024)
void topk_k(const float* __restrict__ probsT, int* __restrict__ sel,
            float* __restrict__ wsel)
{
    __shared__ float vv[4096];
    __shared__ int   ii[4096];
    const int e = blockIdx.x, tid = threadIdx.x;
    for (int i = tid; i < 4096; i += 1024) { vv[i] = probsT[(long long)e * NTOK + i]; ii[i] = i; }
    __syncthreads();
    for (int k = 2; k <= 4096; k <<= 1) {
        for (int j = k >> 1; j > 0; j >>= 1) {
            for (int t = tid; t < 2048; t += 1024) {
                int i = ((t & ~(j - 1)) << 1) | (t & (j - 1));
                int p = i | j;
                bool up = ((i & k) == 0);
                float vi = vv[i], vp = vv[p];
                if ((vi > vp) == up) {
                    vv[i] = vp; vv[p] = vi;
                    int tmp = ii[i]; ii[i] = ii[p]; ii[p] = tmp;
                }
            }
            __syncthreads();
        }
    }
    // ascending sort -> take the top 1024 from the high end (descending)
    sel[e * TOPKn + tid]  = ii[4095 - tid];
    wsel[e * TOPKn + tid] = vv[4095 - tid];
}

__global__ __launch_bounds__(256)
void moe_gather_k(const float* __restrict__ xn, const int* __restrict__ sel,
                  float* __restrict__ xsel)
{
    long long row = blockIdx.x;
    int tok = sel[row];
    const float4* src = (const float4*)(xn + (long long)tok * Dm);
    float4* dst = (float4*)(xsel + row * Dm);
    for (int i = threadIdx.x; i < Dm / 4; i += 256) dst[i] = src[i];
}

__global__ __launch_bounds__(256)
void moe_scatter_k(const float* __restrict__ eout, const int* __restrict__ sel,
                   const float* __restrict__ wsel, float* __restrict__ x)
{
    long long row = blockIdx.x;
    int tok = sel[row];
    float w = wsel[row];
    const float* src = eout + row * Dm;
    float* dst = x + (long long)tok * Dm;
    for (int i = threadIdx.x; i < Dm; i += 256)
        atomicAdd(&dst[i], w * src[i]);
}

// ---------------------------------------------------------------------------
// Host-side orchestration
// ---------------------------------------------------------------------------
extern "C" void kernel_launch(void* const* d_in, const int* in_sizes, int n_in,
                              void* d_out, int out_size, void* d_ws, size_t ws_size,
                              hipStream_t stream)
{
    (void)in_sizes; (void)n_in; (void)out_size; (void)ws_size;

    const int*   idx         = (const int*)  d_in[0];
    const float* wte         = (const float*)d_in[1];
    const float* wpe         = (const float*)d_in[2];
    const float* ln1_w       = (const float*)d_in[3];
    const float* attn_w      = (const float*)d_in[4];
    const float* attn_proj_w = (const float*)d_in[5];
    const float* ln2_w       = (const float*)d_in[6];
    const float* router_w    = (const float*)d_in[7];
    const float* moe_fc_w    = (const float*)d_in[8];
    const float* moe_proj_w  = (const float*)d_in[9];
    const float* lnf_w       = (const float*)d_in[10];

    char* basep = (char*)d_ws;
    size_t off = 0;
    auto alloc = [&](size_t nbytes) -> void* {
        void* p = basep + off;
        off += (nbytes + 255) & ~(size_t)255;
        return p;
    };

    float* x      = (float*)alloc(sizeof(float) * (size_t)NTOK * Dm);
    float* xn     = (float*)alloc(sizeof(float) * (size_t)NTOK * Dm);
    float* qkv    = (float*)alloc(sizeof(float) * (size_t)NTOK * 3 * Dm);
    float* Qb     = (float*)alloc(sizeof(float) * (size_t)NTOK * Dm);
    float* Kb     = (float*)alloc(sizeof(float) * (size_t)NTOK * Dm);
    float* Vtb    = (float*)alloc(sizeof(float) * (size_t)NTOK * Dm);
    float* y      = (float*)alloc(sizeof(float) * (size_t)NTOK * Dm);
    float* rl     = (float*)alloc(sizeof(float) * (size_t)NTOK * En);
    float* probsT = (float*)alloc(sizeof(float) * (size_t)En * NTOK);
    int*   sel    = (int*)  alloc(sizeof(int)   * (size_t)En * TOPKn);
    float* wsel   = (float*)alloc(sizeof(float) * (size_t)En * TOPKn);
    float* xsel   = (float*)alloc(sizeof(float) * (size_t)En * TOPKn * Dm);
    float* hbuf   = (float*)alloc(sizeof(float) * (size_t)En * TOPKn * DFFn);
    float* eout   = (float*)alloc(sizeof(float) * (size_t)En * TOPKn * Dm);
    float* xf     = (float*)alloc(sizeof(float) * (size_t)Bsz * Dm);

    dim3 blk(256);

    embed_k<<<NTOK, blk, 0, stream>>>(idx, wte, wpe, x);

    for (int l = 0; l < Ln; ++l) {
        // --- attention block ---
        rmsnorm_k<<<NTOK, blk, 0, stream>>>(x, Dm, 0, ln1_w + (long long)l * Dm, xn, NTOK);
        gemm_xwt<0, 0><<<dim3(3 * Dm / 128, NTOK / 128, 1), blk, 0, stream>>>(
            xn, 0, attn_w + (long long)l * 3 * Dm * Dm, 0, qkv, 0,
            NTOK, 3 * Dm, Dm, Dm, Dm, 3 * Dm);
        split_qkv_k<<<NTOK, blk, 0, stream>>>(qkv, Qb, Kb, Vtb);
        flash_attn<<<dim3(Tn / 128, Bsz * Hn), blk, 0, stream>>>(Qb, Kb, Vtb, y);
        gemm_xwt<0, 1><<<dim3(Dm / 128, NTOK / 128, 1), blk, 0, stream>>>(   // x += y @ Wp^T
            y, 0, attn_proj_w + (long long)l * Dm * Dm, 0, x, 0,
            NTOK, Dm, Dm, Dm, Dm, Dm);

        // --- MoE block ---
        rmsnorm_k<<<NTOK, blk, 0, stream>>>(x, Dm, 0, ln2_w + (long long)l * Dm, xn, NTOK);
        gemm_xwt<0, 0><<<dim3(1, NTOK / 128, 1), blk, 0, stream>>>(
            xn, 0, router_w + (long long)l * En * Dm, 0, rl, 0,
            NTOK, En, Dm, Dm, Dm, En);
        router_softmax_k<<<NTOK / 256, blk, 0, stream>>>(rl, probsT);
        topk_k<<<En, dim3(1024), 0, stream>>>(probsT, sel, wsel);
        moe_gather_k<<<En * TOPKn, blk, 0, stream>>>(xn, sel, xsel);
        gemm_xwt<1, 0><<<dim3(DFFn / 128, TOPKn / 128, En), blk, 0, stream>>>(  // gelu(xsel@Wfc^T)
            xsel, (long long)TOPKn * Dm,
            moe_fc_w + (long long)l * En * DFFn * Dm, (long long)DFFn * Dm,
            hbuf, (long long)TOPKn * DFFn,
            TOPKn, DFFn, Dm, Dm, Dm, DFFn);
        gemm_xwt<0, 0><<<dim3(Dm / 128, TOPKn / 128, En), blk, 0, stream>>>(
            hbuf, (long long)TOPKn * DFFn,
            moe_proj_w + (long long)l * En * Dm * DFFn, (long long)Dm * DFFn,
            eout, (long long)TOPKn * Dm,
            TOPKn, Dm, DFFn, DFFn, DFFn, Dm);
        moe_scatter_k<<<En * TOPKn, blk, 0, stream>>>(eout, sel, wsel, x);
    }

    // final norm on last position of each batch, then tied lm head
    rmsnorm_k<<<Bsz, blk, 0, stream>>>(x, (long long)Tn * Dm, (long long)(Tn - 1) * Dm,
                                       lnf_w, xf, Bsz);
    gemm_xwt<0, 0><<<dim3((Vn + 127) / 128, 1, 1), blk, 0, stream>>>(
        xf, 0, wte, 0, (float*)d_out, 0, Bsz, Vn, Dm, Dm, Dm, Vn);
}
